// ImprovedPepLand_6305011991144
// MI455X (gfx1250) — compile-verified, matrix operand order checked by reading, not run on hardware
//
#include <hip/hip_runtime.h>
#include <hip/hip_bf16.h>
#include <math.h>

// ---------------- model dims ----------------
#define BB 64
#define NN 128
#define DD 512
#define HH 8
#define LL 12
#define FF 2048
#define TOTAL_N (BB * NN)      // 8192
#define EDGES 131072
#define ATOM_D 42
#define FRAG_D 196
#define BOND_D 14
#define MAX_DEG 100
#define MAX_SP 512
#define DH 64

typedef __attribute__((ext_vector_type(16))) _Float16 v16h;
typedef __attribute__((ext_vector_type(8)))  float    v8f;

// ---- CDNA5 async global->LDS path (guarded so compile never regresses) ----
#if defined(__HIP_DEVICE_COMPILE__) && __has_builtin(__builtin_amdgcn_global_load_async_to_lds_b128)
#define HAVE_ASYNC_LDS 1
#else
#define HAVE_ASYNC_LDS 0
#endif
#if defined(__HIP_DEVICE_COMPILE__) && __has_builtin(__builtin_amdgcn_s_wait_asynccnt)
#define HAVE_WAIT_ASYNC_BUILTIN 1
#else
#define HAVE_WAIT_ASYNC_BUILTIN 0
#endif

#define ASPG __attribute__((address_space(1)))
#define ASPL __attribute__((address_space(3)))

// the builtin takes pointers to 16-byte int vectors (per hipcc diagnostic)
typedef int v4i_b128 __attribute__((vector_size(4 * sizeof(int))));
typedef ASPG v4i_b128* g_b128_p;
typedef ASPL v4i_b128* l_b128_p;

__device__ __forceinline__ void copy16_g2l(float* l, const float* g) {
#if HAVE_ASYNC_LDS
    __builtin_amdgcn_global_load_async_to_lds_b128(
        (g_b128_p)(ASPG void*)(void*)const_cast<float*>(g),
        (l_b128_p)(ASPL void*)(void*)l,
        /*imm offset*/0, /*cpol*/0);
#else
    *(float4*)l = *(const float4*)g;
#endif
}

__device__ __forceinline__ void wait_async_done() {
#if HAVE_ASYNC_LDS
#if HAVE_WAIT_ASYNC_BUILTIN
    __builtin_amdgcn_s_wait_asynccnt(0);
#else
    asm volatile("s_wait_asynccnt 0x0" ::: "memory");
#endif
#endif
}

__device__ __forceinline__ float gelu_f(float x) {
    return 0.5f * x * (1.0f + erff(x * 0.70710678118654752f));
}

__device__ __forceinline__ v8f wmma16(const v16h& a, const v16h& b, v8f c) {
    return __builtin_amdgcn_wmma_f32_16x16x32_f16(false, a, false, b, (short)0, c, false, false);
}

// block-wide sum over 256 threads
__device__ __forceinline__ float block_sum_256(float v, float* sbuf) {
    int tid = threadIdx.x;
    sbuf[tid] = v; __syncthreads();
    for (int off = 128; off > 0; off >>= 1) {
        if (tid < off) sbuf[tid] += sbuf[tid + off];
        __syncthreads();
    }
    float r = sbuf[0]; __syncthreads();
    return r;
}

// =====================================================================
// Generic WMMA GEMM: C[M,N] = act(A[M,K] @ B[K,N] + bias (+ Res))
// block = 128 threads (4 waves), tile 64x64, K-step 32.
// Double-buffered LDS tiles fed by global_load_async_to_lds_b128:
//   wait_async -> barrier -> kick tile k+1 into other buffer -> WMMA tile k
// =====================================================================
#define KT 32
template <int ACT>
__global__ __launch_bounds__(128)
void gemm_wmma(const float* __restrict__ A, const float* __restrict__ Bw,
               const float* __restrict__ bias, const float* __restrict__ Res,
               float* __restrict__ Cout, int M, int Nn, int K) {
    __shared__ __attribute__((aligned(16))) float Asf[2][64 * KT];  // [64][32] f32
    __shared__ __attribute__((aligned(16))) float Bsf[2][KT * 64];  // [32][64] f32
    const int tid  = threadIdx.x;
    const int lane = tid & 31;
    const int bm = blockIdx.y * 64, bn = blockIdx.x * 64;
    const int wave = tid >> 5;
    const int wm = (wave & 1) * 32, wn = (wave >> 1) * 32;

    // stage one K-tile (A: 512 float4 chunks, B: 512 float4 chunks)
    auto issue_tile = [&](int k0, int bufi) {
#pragma unroll
        for (int i = 0; i < 4; ++i) {
            int cid = tid + i * 128;
            int r = cid >> 3, c4 = (cid & 7) * 4;
            copy16_g2l(&Asf[bufi][r * KT + c4], &A[(size_t)(bm + r) * K + k0 + c4]);
        }
#pragma unroll
        for (int i = 0; i < 4; ++i) {
            int cid = tid + i * 128;
            int r = cid >> 4, c4 = (cid & 15) * 4;
            copy16_g2l(&Bsf[bufi][r * 64 + c4], &Bw[(size_t)(k0 + r) * Nn + bn + c4]);
        }
    };

    issue_tile(0, 0);

    v8f acc[2][2] = {};
    int buf = 0;
    for (int k0 = 0; k0 < K; k0 += KT) {
        wait_async_done();      // current tile resident in LDS
        __syncthreads();        // all waves done with the other buffer
        if (k0 + KT < K) issue_tile(k0 + KT, buf ^ 1);  // overlap next tile with math

        // build fragments per documented CDNA5 16-bit layouts (f32 LDS -> f16)
        v16h afr[2], bfr[2];
        const int kb = (lane >> 4) * 8;
#pragma unroll
        for (int i = 0; i < 2; ++i) {
            const float* p = &Asf[buf][(wm + 16 * i + (lane & 15)) * KT];
#pragma unroll
            for (int t = 0; t < 4; ++t) {
                afr[i][2 * t]     = (_Float16)p[kb + 2 * t];
                afr[i][2 * t + 1] = (_Float16)p[kb + 2 * t + 1];
                afr[i][8 + 2 * t]     = (_Float16)p[kb + 16 + 2 * t];
                afr[i][8 + 2 * t + 1] = (_Float16)p[kb + 16 + 2 * t + 1];
            }
        }
#pragma unroll
        for (int j = 0; j < 2; ++j) {
            const int c = wn + 16 * j + (lane & 15);
#pragma unroll
            for (int t = 0; t < 4; ++t) {
                bfr[j][2 * t]     = (_Float16)Bsf[buf][(kb + 2 * t) * 64 + c];
                bfr[j][2 * t + 1] = (_Float16)Bsf[buf][(kb + 2 * t + 1) * 64 + c];
                bfr[j][8 + 2 * t]     = (_Float16)Bsf[buf][(kb + 16 + 2 * t) * 64 + c];
                bfr[j][8 + 2 * t + 1] = (_Float16)Bsf[buf][(kb + 16 + 2 * t + 1) * 64 + c];
            }
        }
#pragma unroll
        for (int i = 0; i < 2; ++i)
#pragma unroll
            for (int j = 0; j < 2; ++j)
                acc[i][j] = wmma16(afr[i], bfr[j], acc[i][j]);
        buf ^= 1;
    }

    // epilogue: C layout lane holds col lane&15; VGPR r -> row r + 8*(lane>>4)
#pragma unroll
    for (int i = 0; i < 2; ++i)
#pragma unroll
        for (int j = 0; j < 2; ++j)
#pragma unroll
            for (int r = 0; r < 8; ++r) {
                int row = bm + wm + 16 * i + r + 8 * (lane >> 4);
                int col = bn + wn + 16 * j + (lane & 15);
                float v = acc[i][j][r];
                if (bias) v += bias[col];
                if (Res)  v += Res[(size_t)row * Nn + col];
                if (ACT == 1) v = gelu_f(v);
                Cout[(size_t)row * Nn + col] = v;
            }
}

// =====================================================================
// Fused per-(b,h) attention: S = Q Kt * scale + spatial bias, softmax, O = P V
// grid = B*H blocks, 256 threads (8 waves); each wave handles 16 query rows
// =====================================================================
__global__ __launch_bounds__(256)
void attn_kernel(const float* __restrict__ Q, const float* __restrict__ Kf,
                 const float* __restrict__ Vf, float* __restrict__ O,
                 const int* __restrict__ sp, const float* __restrict__ spe) {
    __shared__ _Float16 Kt[NN * DH];       // [key][d]  16 KB
    __shared__ _Float16 Vt[NN * DH];       // [key][d]  16 KB
    __shared__ _Float16 Ps[8 * 16 * NN];   // per-wave P rows, 32 KB

    const int b  = blockIdx.x / HH;
    const int hh = blockIdx.x % HH;
    const int tid = threadIdx.x, lane = tid & 31, wave = tid >> 5;
    const float scale = 0.125f;  // 1/sqrt(64)

    // stage K,V head slice into LDS as f16
#pragma unroll
    for (int i = 0; i < 32; ++i) {
        int idx = tid + i * 256;               // 8192 elems
        int n = idx >> 6, d = idx & 63;
        size_t g = (size_t)(b * NN + n) * DD + hh * DH + d;
        Kt[idx] = (_Float16)Kf[g];
        Vt[idx] = (_Float16)Vf[g];
    }
    __syncthreads();

    const int qm = wave * 16;
    const int kb = (lane >> 4) * 8;

    // Q fragments (2 K-steps of 32 over DH=64), gathered straight from global
    v16h qa[2];
#pragma unroll
    for (int s = 0; s < 2; ++s) {
        const float* p = &Q[(size_t)(b * NN + qm + (lane & 15)) * DD + hh * DH + s * 32];
#pragma unroll
        for (int t = 0; t < 4; ++t) {
            qa[s][2 * t]     = (_Float16)p[kb + 2 * t];
            qa[s][2 * t + 1] = (_Float16)p[kb + 2 * t + 1];
            qa[s][8 + 2 * t]     = (_Float16)p[kb + 16 + 2 * t];
            qa[s][8 + 2 * t + 1] = (_Float16)p[kb + 16 + 2 * t + 1];
        }
    }

    // S tiles: 8 key tiles of 16
    v8f sc[8];
#pragma unroll
    for (int j = 0; j < 8; ++j) {
        v8f s = {};
#pragma unroll
        for (int ks = 0; ks < 2; ++ks) {
            v16h bf;
            const int c = j * 16 + (lane & 15);
            const int k0 = kb + ks * 32;
#pragma unroll
            for (int t = 0; t < 4; ++t) {
                bf[2 * t]     = Kt[c * DH + k0 + 2 * t];       // B[d][key] = Kt[key][d]
                bf[2 * t + 1] = Kt[c * DH + k0 + 2 * t + 1];
                bf[8 + 2 * t]     = Kt[c * DH + k0 + 16 + 2 * t];
                bf[8 + 2 * t + 1] = Kt[c * DH + k0 + 16 + 2 * t + 1];
            }
            s = wmma16(qa[ks], bf, s);
        }
        sc[j] = s;
    }

    // scale + spatial-pos bias + softmax (rows live across 16-lane half-waves)
    float rowmax[8], rowsum[8];
#pragma unroll
    for (int r = 0; r < 8; ++r) {
        const int m = r + 8 * (lane >> 4);
        float mx = -1e30f;
#pragma unroll
        for (int j = 0; j < 8; ++j) {
            int key = j * 16 + (lane & 15);
            int spv = sp[(size_t)(b * NN + qm + m) * NN + key];
            spv = spv < 0 ? 0 : (spv > MAX_SP ? MAX_SP : spv);
            float v = sc[j][r] * scale + spe[spv * HH + hh];
            sc[j][r] = v;
            mx = fmaxf(mx, v);
        }
#pragma unroll
        for (int d = 1; d < 16; d <<= 1) mx = fmaxf(mx, __shfl_xor(mx, d, 32));
        rowmax[r] = mx;
    }
#pragma unroll
    for (int r = 0; r < 8; ++r) {
        float s = 0.f;
#pragma unroll
        for (int j = 0; j < 8; ++j) {
            float e = __expf(sc[j][r] - rowmax[r]);
            sc[j][r] = e;
            s += e;
        }
#pragma unroll
        for (int d = 1; d < 16; d <<= 1) s += __shfl_xor(s, d, 32);
        rowsum[r] = s;
    }
    // write P (f16) to this wave's LDS strip
#pragma unroll
    for (int r = 0; r < 8; ++r) {
        const int m = r + 8 * (lane >> 4);
        const float inv = 1.0f / rowsum[r];
#pragma unroll
        for (int j = 0; j < 8; ++j)
            Ps[wave * 16 * NN + m * NN + j * 16 + (lane & 15)] = (_Float16)(sc[j][r] * inv);
    }
    __syncthreads();

    // O = P[16x128] @ V[128x64] : 4 d-tiles, 4 key-steps of 32
#pragma unroll
    for (int dj = 0; dj < 4; ++dj) {
        v8f o = {};
#pragma unroll
        for (int ks = 0; ks < 4; ++ks) {
            v16h pa, vb;
            const _Float16* pp = &Ps[wave * 16 * NN + (lane & 15) * NN + kb + ks * 32];
            const int c = dj * 16 + (lane & 15);
            const int k0 = kb + ks * 32;
#pragma unroll
            for (int t = 0; t < 4; ++t) {
                pa[2 * t]     = pp[2 * t];
                pa[2 * t + 1] = pp[2 * t + 1];
                pa[8 + 2 * t]     = pp[16 + 2 * t];
                pa[8 + 2 * t + 1] = pp[16 + 2 * t + 1];
                vb[2 * t]     = Vt[(k0 + 2 * t) * DH + c];
                vb[2 * t + 1] = Vt[(k0 + 2 * t + 1) * DH + c];
                vb[8 + 2 * t]     = Vt[(k0 + 16 + 2 * t) * DH + c];
                vb[8 + 2 * t + 1] = Vt[(k0 + 16 + 2 * t + 1) * DH + c];
            }
            o = wmma16(pa, vb, o);
        }
#pragma unroll
        for (int r = 0; r < 8; ++r) {
            int m = r + 8 * (lane >> 4);
            O[(size_t)(b * NN + qm + m) * DD + hh * DH + dj * 16 + (lane & 15)] = o[r];
        }
    }
}

// =====================================================================
// Type-routed node encoder: gelu(ln(x @ W + b)) with atom/frag routing
// =====================================================================
__global__ __launch_bounds__(256)
void node_encode(const float* __restrict__ atomf, const float* __restrict__ fragf,
                 const int* __restrict__ ntype,
                 const float* aW, const float* ab, const float* ag, const float* abe,
                 const float* fW, const float* fb, const float* fg, const float* fbe,
                 float* __restrict__ out) {
    __shared__ float xs[FRAG_D];
    __shared__ float row[DD];
    __shared__ float sbuf[256];
    const int n = blockIdx.x, tid = threadIdx.x;
    const int t = ntype[n];
    const float* X; const float* W; const float* bv; const float* g; const float* be; int K;
    if (t == 0) { X = atomf + (size_t)n * ATOM_D; W = aW; bv = ab; g = ag; be = abe; K = ATOM_D; }
    else        { X = fragf + (size_t)n * FRAG_D; W = fW; bv = fb; g = fg; be = fbe; K = FRAG_D; }
    for (int i = tid; i < K; i += 256) xs[i] = X[i];
    __syncthreads();
    for (int c = tid; c < DD; c += 256) {
        float s = bv[c];
        for (int k = 0; k < K; ++k) s += xs[k] * W[(size_t)k * DD + c];
        row[c] = s;
    }
    __syncthreads();
    float mu = block_sum_256(row[tid] + row[tid + 256], sbuf) * (1.0f / DD);
    float d0 = row[tid] - mu, d1 = row[tid + 256] - mu;
    float var = block_sum_256(d0 * d0 + d1 * d1, sbuf) * (1.0f / DD);
    float inv = rsqrtf(var + 1e-5f);
    out[(size_t)n * DD + tid]       = gelu_f(d0 * inv * g[tid] + be[tid]);
    out[(size_t)n * DD + tid + 256] = gelu_f(d1 * inv * g[tid + 256] + be[tid + 256]);
}

// bond encoder for a chunk of edges (K=14)
__global__ __launch_bounds__(256)
void bond_encode(const float* __restrict__ bf, const float* W, const float* bv,
                 const float* g, const float* be, float* __restrict__ out) {
    __shared__ float xs[BOND_D];
    __shared__ float row[DD];
    __shared__ float sbuf[256];
    const int e = blockIdx.x, tid = threadIdx.x;
    if (tid < BOND_D) xs[tid] = bf[(size_t)e * BOND_D + tid];
    __syncthreads();
    for (int c = tid; c < DD; c += 256) {
        float s = bv[c];
#pragma unroll
        for (int k = 0; k < BOND_D; ++k) s += xs[k] * W[k * DD + c];
        row[c] = s;
    }
    __syncthreads();
    float mu = block_sum_256(row[tid] + row[tid + 256], sbuf) * (1.0f / DD);
    float d0 = row[tid] - mu, d1 = row[tid + 256] - mu;
    float var = block_sum_256(d0 * d0 + d1 * d1, sbuf) * (1.0f / DD);
    float inv = rsqrtf(var + 1e-5f);
    out[(size_t)e * DD + tid]       = gelu_f(d0 * inv * g[tid] + be[tid]);
    out[(size_t)e * DD + tid + 256] = gelu_f(d1 * inv * g[tid + 256] + be[tid + 256]);
}

// LayerNorm over rows of width 512 (+optional gelu)
__global__ __launch_bounds__(256)
void ln512(const float* __restrict__ X, const float* g, const float* be,
           float* __restrict__ out, int act) {
    __shared__ float sbuf[256];
    const int row = blockIdx.x, tid = threadIdx.x;
    const float* p = X + (size_t)row * DD;
    float x0 = p[tid], x1 = p[tid + 256];
    float mu = block_sum_256(x0 + x1, sbuf) * (1.0f / DD);
    float d0 = x0 - mu, d1 = x1 - mu;
    float var = block_sum_256(d0 * d0 + d1 * d1, sbuf) * (1.0f / DD);
    float inv = rsqrtf(var + 1e-5f);
    float v0 = d0 * inv * g[tid] + be[tid];
    float v1 = d1 * inv * g[tid + 256] + be[tid + 256];
    if (act) { v0 = gelu_f(v0); v1 = gelu_f(v1); }
    out[(size_t)row * DD + tid] = v0;
    out[(size_t)row * DD + tid + 256] = v1;
}

// degree accumulation
__global__ void deg_kernel(const int* __restrict__ src, const int* __restrict__ dst,
                           float* ind, float* outd) {
    int e = blockIdx.x * 256 + threadIdx.x;
    if (e < EDGES) {
        atomicAdd(&ind[dst[e]], 1.0f);
        atomicAdd(&outd[src[e]], 1.0f);
    }
}

// scatter-add msg chunk into agg by edge destination
__global__ void scatter_kernel(const float* __restrict__ msg, const int* __restrict__ dst,
                               float* __restrict__ agg, int nedge) {
    int gid = blockIdx.x * 256 + threadIdx.x;
    int le = gid >> 9, c = gid & 511;
    if (le < nedge)
        atomicAdd(&agg[(size_t)dst[le] * DD + c], msg[(size_t)le * DD + c]);
}

// per-graph mean over 128 nodes
__global__ __launch_bounds__(256)
void mean_nodes(const float* __restrict__ src, float* __restrict__ dst) {
    int b = blockIdx.x, tid = threadIdx.x;
    for (int c = tid; c < DD; c += 256) {
        float s = 0.f;
        for (int n = 0; n < NN; ++n) s += src[(size_t)(b * NN + n) * DD + c];
        dst[(size_t)b * DD + c] = s * (1.0f / NN);
    }
}

// build vn_in = concat(vn_emb, graph_mean)
__global__ void build_vnin(const float* __restrict__ emb, const float* __restrict__ gm,
                           float* __restrict__ vnin) {
    int b = blockIdx.x, tid = threadIdx.x;
    for (int c = tid; c < 2 * DD; c += 256)
        vnin[(size_t)b * 2 * DD + c] = (c < DD) ? emb[c] : gm[(size_t)b * DD + (c - DD)];
}

// x = node_feat + vn[b] + (indeg>0 ? agg/indeg : 0) + in_emb + out_emb
__global__ __launch_bounds__(256)
void assemble_x(const float* __restrict__ nf, const float* __restrict__ vn,
                const float* __restrict__ agg, const float* __restrict__ ind,
                const float* __restrict__ outd, const float* __restrict__ inemb,
                const float* __restrict__ outemb, float* __restrict__ x) {
    int n = blockIdx.x, tid = threadIdx.x;
    int b = n >> 7;
    float di = ind[n], dout = outd[n];
    int ii = (int)di;   ii = ii < 0 ? 0 : (ii > MAX_DEG ? MAX_DEG : ii);
    int oi = (int)dout; oi = oi < 0 ? 0 : (oi > MAX_DEG ? MAX_DEG : oi);
    float rin = (di > 0.f) ? 1.0f / fmaxf(di, 1.0f) : 0.0f;
#pragma unroll
    for (int s = 0; s < 2; ++s) {
        int c = tid + s * 256;
        float v = nf[(size_t)n * DD + c] + vn[(size_t)b * DD + c]
                + agg[(size_t)n * DD + c] * rin
                + inemb[(size_t)ii * DD + c] + outemb[(size_t)oi * DD + c];
        x[(size_t)n * DD + c] = v;
    }
}

// graph_repr = sum_l softmax(layer_w)[l] * means[l]
__global__ void pool_kernel(const float* __restrict__ means, const float* __restrict__ lw,
                            float* __restrict__ repr) {
    int b = blockIdx.x, tid = threadIdx.x;
    float mx = -1e30f;
    for (int l = 0; l < LL; ++l) mx = fmaxf(mx, lw[l]);
    float den = 0.f, w[LL];
    for (int l = 0; l < LL; ++l) { w[l] = __expf(lw[l] - mx); den += w[l]; }
    float invd = 1.0f / den;
#pragma unroll
    for (int s = 0; s < 2; ++s) {
        int c = tid + s * 256;
        float acc = 0.f;
        for (int l = 0; l < LL; ++l)
            acc += w[l] * means[(size_t)l * BB * DD + (size_t)b * DD + c];
        repr[(size_t)b * DD + c] = acc * invd;
    }
}

// =====================================================================
// host orchestration
// =====================================================================
static inline void launch_gemm(int act, const float* A, const float* Bw, const float* bias,
                               const float* Res, float* C, int M, int N, int K,
                               hipStream_t s) {
    dim3 g(N / 64, M / 64), b(128);
    if (act) hipLaunchKernelGGL((gemm_wmma<1>), g, b, 0, s, A, Bw, bias, Res, C, M, N, K);
    else     hipLaunchKernelGGL((gemm_wmma<0>), g, b, 0, s, A, Bw, bias, Res, C, M, N, K);
}

extern "C" void kernel_launch(void* const* d_in, const int* in_sizes, int n_in,
                              void* d_out, int out_size, void* d_ws, size_t ws_size,
                              hipStream_t stream) {
    // ---- inputs in setup_inputs() dict insertion order ----
    const float* atom_feat = (const float*)d_in[0];
    const float* frag_feat = (const float*)d_in[1];
    const float* bond_feat = (const float*)d_in[2];
    const int*   node_type = (const int*)d_in[3];
    const int*   edge_src  = (const int*)d_in[4];
    const int*   edge_dst  = (const int*)d_in[5];
    const int*   spatial   = (const int*)d_in[6];
    int i = 7;
    const float *atom_W = (const float*)d_in[i++], *atom_b = (const float*)d_in[i++],
                *atom_g = (const float*)d_in[i++], *atom_be = (const float*)d_in[i++];
    const float *frag_W = (const float*)d_in[i++], *frag_b = (const float*)d_in[i++],
                *frag_g = (const float*)d_in[i++], *frag_be = (const float*)d_in[i++];
    const float *bond_W = (const float*)d_in[i++], *bond_b = (const float*)d_in[i++],
                *bond_g = (const float*)d_in[i++], *bond_be = (const float*)d_in[i++];
    const float *vn_emb = (const float*)d_in[i++];
    const float *vn_W1 = (const float*)d_in[i++], *vn_b1 = (const float*)d_in[i++];
    const float *vn_g = (const float*)d_in[i++],  *vn_be = (const float*)d_in[i++];
    const float *vn_W2 = (const float*)d_in[i++], *vn_b2 = (const float*)d_in[i++];
    const float *e2n_W = (const float*)d_in[i++], *e2n_b = (const float*)d_in[i++];
    const float *in_emb = (const float*)d_in[i++], *out_emb = (const float*)d_in[i++];
    const float *sp_emb = (const float*)d_in[i++];
    const float *ln1_g = (const float*)d_in[i++], *ln1_b = (const float*)d_in[i++];
    const float *ln2_g = (const float*)d_in[i++], *ln2_b = (const float*)d_in[i++];
    const float *Wq = (const float*)d_in[i++], *bq = (const float*)d_in[i++];
    const float *Wk = (const float*)d_in[i++], *bk = (const float*)d_in[i++];
    const float *Wv = (const float*)d_in[i++], *bv = (const float*)d_in[i++];
    const float *Wo = (const float*)d_in[i++], *bo = (const float*)d_in[i++];
    const float *Wf1 = (const float*)d_in[i++], *bf1 = (const float*)d_in[i++];
    const float *Wf2 = (const float*)d_in[i++], *bf2 = (const float*)d_in[i++];
    const float *layer_w = (const float*)d_in[i++];
    const float *out_W = (const float*)d_in[i++], *out_b = (const float*)d_in[i++];
    const float *out_g = (const float*)d_in[i++], *out_be = (const float*)d_in[i++];
    (void)in_sizes; (void)n_in; (void)out_size; (void)ws_size;

    // ---- workspace carve-out ----
    const size_t SZ = (size_t)TOTAL_N * DD;  // 4,194,304 floats
    float* w = (float*)d_ws;
    float* nf   = w;            w += SZ;
    float* x    = w;            w += SZ;
    float* hbuf = w;            w += SZ;
    float* qbuf = w;            w += SZ;
    float* kbuf = w;            w += SZ;
    float* vbuf = w;            w += SZ;
    float* obuf = w;            w += SZ;
    float* agg  = w;            w += SZ;
    float* encb = w;            w += SZ;   // edge-chunk encoder out (8192x512)
    float* msgb = w;            w += SZ;   // edge-chunk msg (8192x512)
    float* ffb  = w;            w += (size_t)TOTAL_N * FF;
    float* means = w;           w += (size_t)LL * BB * DD;
    float* gmean = w;           w += (size_t)BB * DD;
    float* vnin  = w;           w += (size_t)BB * 2 * DD;
    float* vnh   = w;           w += (size_t)BB * DD;
    float* vnb   = w;           w += (size_t)BB * DD;
    float* reprb = w;           w += (size_t)BB * DD;
    float* tmpb  = w;           w += (size_t)BB * DD;
    float* indeg = w;           w += TOTAL_N;
    float* outdeg = w;          w += TOTAL_N;

    (void)hipMemsetAsync(agg, 0, SZ * sizeof(float), stream);
    (void)hipMemsetAsync(indeg, 0, TOTAL_N * sizeof(float), stream);
    (void)hipMemsetAsync(outdeg, 0, TOTAL_N * sizeof(float), stream);

    // ---- encoders + virtual node ----
    node_encode<<<TOTAL_N, 256, 0, stream>>>(atom_feat, frag_feat, node_type,
        atom_W, atom_b, atom_g, atom_be, frag_W, frag_b, frag_g, frag_be, nf);
    mean_nodes<<<BB, 256, 0, stream>>>(nf, gmean);
    build_vnin<<<BB, 256, 0, stream>>>(vn_emb, gmean, vnin);
    launch_gemm(1, vnin, vn_W1, vn_b1, nullptr, vnh, BB, DD, 2 * DD, stream);  // gelu
    ln512<<<BB, 256, 0, stream>>>(vnh, vn_g, vn_be, vnh, 0);
    launch_gemm(0, vnh, vn_W2, vn_b2, nullptr, vnb, BB, DD, DD, stream);

    // ---- degrees + edge->node aggregation (chunked) ----
    deg_kernel<<<EDGES / 256, 256, 0, stream>>>(edge_src, edge_dst, indeg, outdeg);
    const int ECH = 8192;
    for (int e0 = 0; e0 < EDGES; e0 += ECH) {
        bond_encode<<<ECH, 256, 0, stream>>>(bond_feat + (size_t)e0 * BOND_D,
                                             bond_W, bond_b, bond_g, bond_be, encb);
        launch_gemm(0, encb, e2n_W, e2n_b, nullptr, msgb, ECH, DD, DD, stream);
        scatter_kernel<<<(ECH * DD) / 256, 256, 0, stream>>>(msgb, edge_dst + e0, agg, ECH);
    }

    assemble_x<<<TOTAL_N, 256, 0, stream>>>(nf, vnb, agg, indeg, outdeg, in_emb, out_emb, x);

    // ---- transformer layers ----
    for (int l = 0; l < LL; ++l) {
        const size_t oD = (size_t)l * DD, oDD = (size_t)l * DD * DD;
        ln512<<<TOTAL_N, 256, 0, stream>>>(x, ln1_g + oD, ln1_b + oD, hbuf, 0);
        launch_gemm(0, hbuf, Wq + oDD, bq + oD, nullptr, qbuf, TOTAL_N, DD, DD, stream);
        launch_gemm(0, hbuf, Wk + oDD, bk + oD, nullptr, kbuf, TOTAL_N, DD, DD, stream);
        launch_gemm(0, hbuf, Wv + oDD, bv + oD, nullptr, vbuf, TOTAL_N, DD, DD, stream);
        attn_kernel<<<BB * HH, 256, 0, stream>>>(qbuf, kbuf, vbuf, obuf, spatial, sp_emb);
        launch_gemm(0, obuf, Wo + oDD, bo + oD, x, x, TOTAL_N, DD, DD, stream);
        ln512<<<TOTAL_N, 256, 0, stream>>>(x, ln2_g + oD, ln2_b + oD, hbuf, 0);
        launch_gemm(1, hbuf, Wf1 + (size_t)l * DD * FF, bf1 + (size_t)l * FF,
                    nullptr, ffb, TOTAL_N, FF, DD, stream);
        launch_gemm(0, ffb, Wf2 + (size_t)l * FF * DD, bf2 + oD, x, x, TOTAL_N, DD, FF, stream);
        mean_nodes<<<BB, 256, 0, stream>>>(x, means + (size_t)l * BB * DD);
    }

    // ---- pooling + output head ----
    pool_kernel<<<BB, 256, 0, stream>>>(means, layer_w, reprb);
    launch_gemm(0, reprb, out_W, out_b, nullptr, tmpb, BB, DD, DD, stream);
    ln512<<<BB, 256, 0, stream>>>(tmpb, out_g, out_be, (float*)d_out, 1);
}
